// MS_SSIM_21320217658065
// MI455X (gfx1250) — matrix operand order repaired and checked by reading, not compiled, not run on hardware
//
#include <hip/hip_runtime.h>
#include <math.h>

// MS-SSIM for X,Y: [16,3,512,512] f32 -> scalar f32.
// Bandwidth-bound: one fused pass per pyramid level. The separable 11-tap
// Gaussian blur of each 16x16 output tile is computed as
//   Out(16x16) = Wv(16x32) * [ S(32x32) * Wh(32x16) ]
// with v_wmma_f32_16x16x32_f16 (f16 operands, f32 accumulate).
// All operand builds are branch-free: the LDS tile is fully initialized over
// the padded 48x48 region (data or 0, never NaN) and the banded Gaussian
// matrices are zero wherever padded values could leak in.

typedef __attribute__((ext_vector_type(16))) _Float16 v16h;
typedef __attribute__((ext_vector_type(8)))  float    v8f;

#define TP 52   // sX/sY row pitch (floats): %4==0 -> 16B-aligned float4 rows
#define SP 36   // T' scratch row pitch (floats): %4==0, bank-spreading pad

// ---------------------------------------------------------------- zero acc
__global__ void zero_acc_kernel(float* __restrict__ acc) {
  if (threadIdx.x < 480) acc[threadIdx.x] = 0.0f;   // 5 levels * 48 planes * 2
}

// ---------------------------------------------------------------- avg pool 2x2
__global__ __launch_bounds__(256) void avgpool_kernel(
    const float* __restrict__ Xi, const float* __restrict__ Yi,
    float* __restrict__ Xo, float* __restrict__ Yo, int Ho) {
  const int Wo = Ho;
  const long n = (long)48 * Ho * Wo;
  long idx = (long)blockIdx.x * blockDim.x + threadIdx.x;
  if (idx >= n) return;
  int wo = (int)(idx % Wo);
  long t = idx / Wo;
  int ho = (int)(t % Ho);
  int pl = (int)(t / Ho);
  int Hi = Ho * 2;
  long base = (long)pl * Hi * Hi + (long)(2 * ho) * Hi + 2 * wo;
  const float2 x0 = *(const float2*)&Xi[base];
  const float2 x1 = *(const float2*)&Xi[base + Hi];
  const float2 y0 = *(const float2*)&Yi[base];
  const float2 y1 = *(const float2*)&Yi[base + Hi];
  Xo[idx] = 0.25f * (x0.x + x0.y + x1.x + x1.y);
  Yo[idx] = 0.25f * (y0.x + y0.y + y1.x + y1.y);
}

// ---------------------------------------------------------------- fused level
// One workgroup = 4 waves; each wave owns one 16x16 output tile of a 32x32
// output region of plane blockIdx.z. All five blurred moments, the SSIM/CS
// map and the per-plane reduction are fused (inputs read exactly once).
__global__ __launch_bounds__(128) void msssim_level_kernel(
    const float* __restrict__ X, const float* __restrict__ Y,
    int H, float* __restrict__ acc) {
  const int W = H;
  const int Hout = H - 10, Wout = W - 10;

  __shared__ float sX[48 * TP];
  __shared__ float sY[48 * TP];
  __shared__ float sT[4][16 * SP];     // per-wave T' scratch: T'[n][k], f32

  const int tid   = threadIdx.x;
  const int lane  = tid & 31;
  const int wave  = tid >> 5;
  const int lhalf = lane >> 4;         // 0: lanes 0-15, 1: lanes 16-31
  const int lm    = lane & 15;

  const int p   = blockIdx.z;          // plane = b*3 + c
  const long plane_off = (long)p * H * W;
  const int gy0 = blockIdx.y * 32;
  const int gx0 = blockIdx.x * 32;

  // Cooperative halo load: full padded 48x48 region, zero past the plane.
  for (int i = tid; i < 48 * 48; i += 128) {
    int r = i / 48, c = i % 48;
    int gr = gy0 + r, gc = gx0 + c;
    bool inb = (gr < H) && (gc < W);
    long idx = plane_off + (long)gr * W + gc;
    sX[r * TP + c] = inb ? X[idx] : 0.0f;
    sY[r * TP + c] = inb ? Y[idx] : 0.0f;
  }

  // Banded Gaussian operands, computed arithmetically (selects, no branches).
  // A-layout (wvA) element i of lane: M = lane%16, K = (lane/16)*16 + i
  // B-layout (whB) element i of lane: N = lane%16, K = (i/8)*16 + (lane/16)*8 + i%8
  const float inv2s2 = 1.0f / (2.0f * 1.5f * 1.5f);
  float zsum = 0.0f;
#pragma unroll
  for (int k = 0; k < 11; ++k) {
    float ck = (float)k - 5.0f;
    zsum += __expf(-(ck * ck) * inv2s2);
  }
  const float invZ = 1.0f / zsum;
  v16h wvA;   // Wv[M][K] = g[K-M]
  v16h whB;   // Wh[K][N] = g[K-N]
#pragma unroll
  for (int i = 0; i < 16; ++i) {
    int kA = (lhalf << 4) + i;
    int dA = kA - lm;
    float tA = (float)dA - 5.0f;
    float eA = __expf(-(tA * tA) * inv2s2) * invZ;
    wvA[i] = (_Float16)((dA >= 0 && dA <= 10) ? eA : 0.0f);
    int kB = ((i >> 3) << 4) + (lhalf << 3) + (i & 7);
    int dB = kB - lm;
    float tB = (float)dB - 5.0f;
    float eB = __expf(-(tB * tB) * inv2s2) * invZ;
    whB[i] = (_Float16)((dB >= 0 && dB <= 10) ? eB : 0.0f);
  }
  __syncthreads();

  const int by = (wave >> 1) << 4;     // tile origin inside the 48x48 region
  const int bx = (wave & 1) << 4;
  float* Tld = &sT[wave][0];
  const v8f zeroC = {};

  // Hoist all tile reads: each lane needs cols [c0..c0+7] and [c1..c1+7]
  // of its two A rows -> aligned float4 loads, kept in registers.
  const int c0 = bx + (lhalf << 3);
  const int c1 = c0 + 16;
  float xr[2][16], yr[2][16];
#pragma unroll
  for (int blk = 0; blk < 2; ++blk) {
    int r = by + (blk << 4) + lm;
    const float* px = &sX[r * TP];
    const float* py = &sY[r * TP];
    float4 xv0 = *(const float4*)(px + c0);
    float4 xv1 = *(const float4*)(px + c0 + 4);
    float4 xv2 = *(const float4*)(px + c1);
    float4 xv3 = *(const float4*)(px + c1 + 4);
    float4 yv0 = *(const float4*)(py + c0);
    float4 yv1 = *(const float4*)(py + c0 + 4);
    float4 yv2 = *(const float4*)(py + c1);
    float4 yv3 = *(const float4*)(py + c1 + 4);
    xr[blk][0] = xv0.x;  xr[blk][1] = xv0.y;  xr[blk][2]  = xv0.z;  xr[blk][3]  = xv0.w;
    xr[blk][4] = xv1.x;  xr[blk][5] = xv1.y;  xr[blk][6]  = xv1.z;  xr[blk][7]  = xv1.w;
    xr[blk][8] = xv2.x;  xr[blk][9] = xv2.y;  xr[blk][10] = xv2.z;  xr[blk][11] = xv2.w;
    xr[blk][12] = xv3.x; xr[blk][13] = xv3.y; xr[blk][14] = xv3.z;  xr[blk][15] = xv3.w;
    yr[blk][0] = yv0.x;  yr[blk][1] = yv0.y;  yr[blk][2]  = yv0.z;  yr[blk][3]  = yv0.w;
    yr[blk][4] = yv1.x;  yr[blk][5] = yv1.y;  yr[blk][6]  = yv1.z;  yr[blk][7]  = yv1.w;
    yr[blk][8] = yv2.x;  yr[blk][9] = yv2.y;  yr[blk][10] = yv2.z;  yr[blk][11] = yv2.w;
    yr[blk][12] = yv3.x; yr[blk][13] = yv3.y; yr[blk][14] = yv3.z;  yr[blk][15] = yv3.w;
  }

  v8f blur[5];                          // mu1, mu2, E[X^2], E[Y^2], E[XY]
#pragma unroll
  for (int q = 0; q < 5; ++q) {
    v8f d01[2];
#pragma unroll
    for (int blk = 0; blk < 2; ++blk) {
      v16h a;
#pragma unroll
      for (int i = 0; i < 16; ++i) {
        float xv = xr[blk][i], yv = yr[blk][i];
        float v = (q == 0) ? xv
                : (q == 1) ? yv
                : (q == 2) ? xv * xv
                : (q == 3) ? yv * yv
                           : xv * yv;
        a[i] = (_Float16)v;
      }
      // T rows [blk*16 .. blk*16+15] = S_rows * Wh  (horizontal pass)
      d01[blk] = __builtin_amdgcn_wmma_f32_16x16x32_f16(
          false, a, false, whB, (short)0, zeroC, false, false);
    }
    // D layout -> LDS scratch as T'[n][k]: per lane 8 consecutive floats.
#pragma unroll
    for (int v = 0; v < 8; ++v) {
      Tld[lm * SP + (lhalf << 3) + v]      = d01[0][v];
      Tld[lm * SP + 16 + (lhalf << 3) + v] = d01[1][v];
    }
    asm volatile("s_wait_dscnt 0" ::: "memory");   // per-wave RAW on scratch
    // B-layout regather: K = (lane/16)*16 + i, N = lm -> 16 consecutive f32.
    v16h bT;
#pragma unroll
    for (int i = 0; i < 16; ++i)
      bT[i] = (_Float16)Tld[lm * SP + (lhalf << 4) + i];
    // Out = Wv * T  (vertical pass)
    blur[q] = __builtin_amdgcn_wmma_f32_16x16x32_f16(
        false, wvA, false, bT, (short)0, zeroC, false, false);
  }

  // SSIM / CS map + masked accumulation (selects, no EXEC churn).
  const float C1 = 0.01f * 0.01f;   // (K1*DATA_RANGE)^2
  const float C2 = 0.03f * 0.03f;   // (K2*DATA_RANGE)^2
  float ssim_sum = 0.0f, cs_sum = 0.0f;
#pragma unroll
  for (int v = 0; v < 8; ++v) {
    int m   = v + (lhalf << 3);
    int row = gy0 + by + m;
    int col = gx0 + bx + lm;
    float mu1 = blur[0][v], mu2 = blur[1][v];
    float mu1s = mu1 * mu1, mu2s = mu2 * mu2, mu12 = mu1 * mu2;
    float s1  = blur[2][v] - mu1s;
    float s2  = blur[3][v] - mu2s;
    float s12 = blur[4][v] - mu12;
    float cs = (2.0f * s12 + C2) / (s1 + s2 + C2);
    float ss = ((2.0f * mu12 + C1) / (mu1s + mu2s + C1)) * cs;
    bool valid = (row < Hout) && (col < Wout);
    ssim_sum += valid ? ss : 0.0f;
    cs_sum   += valid ? cs : 0.0f;
  }
  // wave32 tree reduction
#pragma unroll
  for (int off = 16; off >= 1; off >>= 1) {
    ssim_sum += __shfl_xor(ssim_sum, off, 32);
    cs_sum   += __shfl_xor(cs_sum, off, 32);
  }
  if (lane == 0) {
    atomicAdd(&acc[p * 2 + 0], ssim_sum);
    atomicAdd(&acc[p * 2 + 1], cs_sum);
  }
}

// ---------------------------------------------------------------- finalize
__global__ void finalize_kernel(const float* __restrict__ acc,
                                float* __restrict__ out) {
  __shared__ float red[48];
  const int tid = threadIdx.x;
  const float w[5] = {0.0448f, 0.2856f, 0.3001f, 0.2363f, 0.1333f};
  if (tid < 48) {
    float ms = 1.0f;
#pragma unroll
    for (int l = 0; l < 5; ++l) {
      int Hl = 512 >> l;
      float cnt = (float)((Hl - 10) * (Hl - 10));
      float ssim = acc[l * 96 + tid * 2 + 0] / cnt;
      float cs   = acc[l * 96 + tid * 2 + 1] / cnt;
      float v = (l == 4) ? ssim : cs;
      v = fmaxf(v, 0.0f);
      ms *= __powf(v, w[l]);
    }
    red[tid] = ms;
  }
  __syncthreads();
  if (tid == 0) {
    float s = 0.0f;
    for (int i = 0; i < 48; ++i) s += red[i];
    *out = s / 48.0f;
  }
}

// ---------------------------------------------------------------- launcher
extern "C" void kernel_launch(void* const* d_in, const int* in_sizes, int n_in,
                              void* d_out, int out_size, void* d_ws, size_t ws_size,
                              hipStream_t stream) {
  (void)in_sizes; (void)n_in; (void)out_size; (void)ws_size;
  const float* X0 = (const float*)d_in[0];
  const float* Y0 = (const float*)d_in[1];
  float* ws  = (float*)d_ws;
  float* acc = ws;                 // 512 floats reserved
  size_t off = 512;

  // Pyramid (levels 1..4) lives in workspace: 2 * 4.18M floats ~= 33.4 MB.
  float* Xl[5]; float* Yl[5];
  Xl[0] = (float*)X0; Yl[0] = (float*)Y0;
  int H = 512;
  for (int l = 1; l < 5; ++l) {
    H >>= 1;
    size_t n = (size_t)48 * H * H;
    Xl[l] = ws + off; off += n;
    Yl[l] = ws + off; off += n;
  }

  zero_acc_kernel<<<1, 512, 0, stream>>>(acc);

  H = 512;
  for (int l = 0; l < 4; ++l) {
    int Ho = H >> 1;
    long n = (long)48 * Ho * Ho;
    int blocks = (int)((n + 255) / 256);
    avgpool_kernel<<<blocks, 256, 0, stream>>>(Xl[l], Yl[l], Xl[l + 1], Yl[l + 1], Ho);
    H = Ho;
  }

  H = 512;
  for (int l = 0; l < 5; ++l) {
    int Hout = H - 10;
    int nb = (Hout + 31) / 32;
    msssim_level_kernel<<<dim3(nb, nb, 48), 128, 0, stream>>>(
        Xl[l], Yl[l], H, acc + l * 96);
    H >>= 1;
  }

  finalize_kernel<<<1, 64, 0, stream>>>(acc, (float*)d_out);
}
